// RelationalMemory_64613488001029
// MI455X (gfx1250) — matrix operand order repaired and checked by reading, not compiled, not run on hardware
//
#include <hip/hip_runtime.h>

// ---------------------------------------------------------------------------
// RelationalMemory.recall for MI455X (gfx1250, wave32, WMMA).
//
//   query = l2norm(latent)           [N=16384, D=256]
//   nkeys = l2norm(keys)             [M=8192,  D=256]
//   S     = (query @ nkeys^T) * hardness / 0.1
//   out   = softmax(S, axis=-1) @ vals
//
// Flash-attention-style fused kernel, f16 WMMA / f32 accumulate:
//  * hardness * 10 * log2(e) folded into f16 keys -> softmax is plain exp2.
//  * S^T = K @ Q^T so each lane owns ONE query column: online-softmax
//    max/sum are one scalar/lane (single shfl_xor(16) merge) and P^T
//    converts to the 2nd GEMM's B-fragment entirely in registers.
//  * O^T = V^T @ P^T with V^T staged in padded LDS (conflict-free b128 reads)
//  * K/V^T tiles DOUBLE-BUFFERED via CDNA5 global_load_async_to_lds_b128
//    (ASYNCcnt), prefetching tile kv+1 behind tile kv's 1024 WMMAs.
//  * A-fragments flow through an explicit rotating 4-slot / distance-3
//    software pipeline so ds_load latency overlaps WMMA deterministically.
// ---------------------------------------------------------------------------

typedef __attribute__((ext_vector_type(16))) _Float16 v16h;
typedef __attribute__((ext_vector_type(8)))  float    v8f;

struct alignas(16) U4 { unsigned int a, b, c, d; };
struct alignas(16) F4 { float x, y, z, w; };

union AFrag { U4 u4[2]; v16h h; };            // 16x32 f16 A-matrix fragment
union BFrag { unsigned int u[8]; v16h h; };   // 32x16 f16 B-matrix fragment

static __device__ __forceinline__ v8f zero8() {
  v8f z = {0.f, 0.f, 0.f, 0.f, 0.f, 0.f, 0.f, 0.f};
  return z;
}

static __device__ __forceinline__ unsigned int packh2(float a, float b) {
  return __builtin_bit_cast(unsigned int, __builtin_amdgcn_cvt_pkrtz(a, b));
}

static __device__ __forceinline__ void ld_frag(AFrag& a, const char* p) {
  a.u4[0] = *(const U4*)p;
  a.u4[1] = *(const U4*)(p + 32);
}

// --------------------------- problem constants -----------------------------
#define D_DIM   256
#define M_KEYS  8192
#define N_ROWS  16384          // B*S = 8*2048
#define BM      128            // query rows per workgroup
#define BN      128            // key rows per KV iteration
#define KV_ITER (M_KEYS / BN)  // 64

// LDS: two buffers of {K tile, V^T tile}; rows padded so the 16-lane
// row-strided ds_read_b128 A-fragment loads hit distinct banks.
#define KROW_B  528              // (256+8 pad) halfs * 2B ; 528/4 % 64 == 4
#define KTILE_B (BM * KROW_B)    // 67584
#define VROW_B  272              // (128+8 pad) halfs * 2B ; 272/4 % 64 == 4
#define VTILE_B (D_DIM * VROW_B) // 69632
#define BUF_B   (KTILE_B + VTILE_B)  // 137216
#define SMEM_MAIN (2 * BUF_B)        // 274432 (< 320KB WGP LDS)
#define OSTRIDE 260              // f32 epilogue staging row stride (pad 4)

// ---- CDNA5 async global->LDS copy (ASYNCcnt), inline asm (portable) -------
static __device__ __forceinline__ void async_ld_b128(unsigned lds_off,
                                                     unsigned gvoff,
                                                     const void* sbase) {
  asm volatile("global_load_async_to_lds_b128 %0, %1, %2"
               :: "v"(lds_off), "v"(gvoff), "s"(sbase)
               : "memory");
}
static __device__ __forceinline__ void wait_async_le32() {
  asm volatile("s_wait_asynccnt 0x20" ::: "memory");
}
static __device__ __forceinline__ void wait_async_0() {
  asm volatile("s_wait_asynccnt 0x0" ::: "memory");
}

// Issue the 32 async b128 loads (per thread) staging K and V^T tiles for
// KV block n0 into buffer `buf`.
static __device__ __forceinline__ void stage_tiles(
    unsigned lds0, int buf, const _Float16* __restrict__ K16,
    const _Float16* __restrict__ Vt16, int n0, int tid) {
  {
    const char* gk = (const char*)K16 + (size_t)n0 * (D_DIM * 2);
    const unsigned kb = lds0 + (unsigned)buf * BUF_B;
    const int r = tid >> 5, co = (tid & 31) * 16;  // 128 rows x 512B
#pragma unroll
    for (int i = 0; i < 16; ++i) {
      int row = i * 8 + r;
      async_ld_b128(kb + (unsigned)(row * KROW_B + co),
                    (unsigned)(row * (D_DIM * 2) + co), gk);
    }
  }
  {
    const char* gv = (const char*)Vt16 + (size_t)n0 * 2;
    const unsigned vb = lds0 + (unsigned)buf * BUF_B + KTILE_B;
    const int r = tid >> 4, co = (tid & 15) * 16;  // 256 rows x 256B window
#pragma unroll
    for (int i = 0; i < 16; ++i) {
      int row = i * 16 + r;
      async_ld_b128(vb + (unsigned)(row * VROW_B + co),
                    (unsigned)(row * (M_KEYS * 2) + co), gv);
    }
  }
}

// --------------------------- prep: row l2-normalize ------------------------
__global__ __launch_bounds__(256) void rownorm_f16_kernel(
    const float* __restrict__ in, const float* __restrict__ hard,
    float factor, _Float16* __restrict__ outp) {
  const int row = blockIdx.x;
  const int tid = threadIdx.x;
  float x = in[(size_t)row * D_DIM + tid];
  float s = x * x;
#pragma unroll
  for (int o = 16; o >= 1; o >>= 1) s += __shfl_xor(s, o);
  __shared__ float red[8];
  if ((tid & 31) == 0) red[tid >> 5] = s;
  __syncthreads();
  float tot = red[0] + red[1] + red[2] + red[3] +
              red[4] + red[5] + red[6] + red[7];
  float k = factor / fmaxf(sqrtf(tot), 1e-12f);
  if (hard) k *= hard[row];
  outp[(size_t)row * D_DIM + tid] = (_Float16)(x * k);
}

// --------------------------- prep: vals transpose --------------------------
__global__ __launch_bounds__(256) void transpose_vals_kernel(
    const float* __restrict__ vals, _Float16* __restrict__ vt) {
  __shared__ _Float16 tile[32][33];
  const int tx = threadIdx.x, ty = threadIdx.y;
  const int m0 = blockIdx.x * 32, d0 = blockIdx.y * 32;
#pragma unroll
  for (int i = 0; i < 4; ++i) {
    int m = ty + i * 8;
    tile[m][tx] = (_Float16)vals[(size_t)(m0 + m) * D_DIM + d0 + tx];
  }
  __syncthreads();
#pragma unroll
  for (int i = 0; i < 4; ++i) {
    int d = ty + i * 8;
    vt[(size_t)(d0 + d) * M_KEYS + m0 + tx] = tile[tx][d];
  }
}

// --------------------------- main fused flash kernel -----------------------
__global__ __launch_bounds__(256, 1) void relmem_flash_kernel(
    const _Float16* __restrict__ Q16,   // [N][256]  l2-normalized queries
    const _Float16* __restrict__ K16,   // [M][256]  nkeys * hardness*10*log2e
    const _Float16* __restrict__ Vt16,  // [256][M]  vals^T
    float* __restrict__ out) {          // [N][256]
  extern __shared__ char smem[];
  const unsigned lds0 = (unsigned)(uintptr_t)smem;  // group-relative LDS base

  const int tid  = threadIdx.x;
  const int lane = tid & 31;
  const int wave = tid >> 5;    // 0..7
  const int hi   = lane >> 4;   // half-wave id
  const int ln   = lane & 15;

  // ---- Q B-fragments: loaded once from global, resident all kernel. ------
  v16h qf[8];
  {
    const char* qrow = (const char*)Q16 +
        ((size_t)blockIdx.x * BM + wave * 16 + ln) * (D_DIM * 2) + hi * 16;
#pragma unroll
    for (int c = 0; c < 8; ++c) {
      AFrag t;
      t.u4[0] = *(const U4*)(qrow + c * 64);
      t.u4[1] = *(const U4*)(qrow + c * 64 + 32);
      qf[c] = t.h;
    }
  }

  v8f ot[16];                       // O^T accumulators (16 d-tiles)
#pragma unroll
  for (int dt = 0; dt < 16; ++dt) ot[dt] = zero8();
  float mrun = -3.0e38f;            // running max (exp2 domain), per lane
  float lrun = 0.0f;                // running sum

  // prologue: stage tile 0 into buffer 0
  stage_tiles(lds0, 0, K16, Vt16, 0, tid);

#pragma unroll 1
  for (int kv = 0; kv < KV_ITER; ++kv) {
    const int cur = kv & 1;
    // prefetch next tile into other buffer, then wait for current tile.
    if (kv + 1 < KV_ITER) {
      stage_tiles(lds0, cur ^ 1, K16, Vt16, (kv + 1) * BN, tid);
      wait_async_le32();  // in-order: current tile's 32 loads complete
    } else {
      wait_async_0();
    }
    __syncthreads();  // publish current tile to all waves

    const char* klds = smem + cur * BUF_B;
    const char* vlds = klds + KTILE_B;

    // ---- GEMM1: S^T[n][q] += K[n][d] * Q[q][d] -------------------------
    // 64 steps (c-major, t inner); rotating 4-slot pipeline, distance 3.
    v8f st[8];
#pragma unroll
    for (int t = 0; t < 8; ++t) st[t] = zero8();
    {
      AFrag a[4];
#pragma unroll
      for (int s = 0; s < 3; ++s)
        ld_frag(a[s], klds + ((s & 7) * 16 + ln) * KROW_B +
                      (s >> 3) * 64 + hi * 16);
#pragma unroll
      for (int s = 0; s < 64; ++s) {
        if (s + 3 < 64) {
          int sn = s + 3;
          ld_frag(a[sn & 3], klds + ((sn & 7) * 16 + ln) * KROW_B +
                            (sn >> 3) * 64 + hi * 16);
        }
        const int c = s >> 3, t = s & 7;
        st[t] = __builtin_amdgcn_wmma_f32_16x16x32_f16(
            false, a[s & 3].h, false, qf[c], (short)0, st[t], false, false);
      }
    }

    // ---- online softmax (exp2 domain; all scaling folded into K16) ----
    float mc = -3.0e38f;
#pragma unroll
    for (int t = 0; t < 8; ++t)
#pragma unroll
      for (int r = 0; r < 8; ++r) mc = fmaxf(mc, st[t][r]);
    mc = fmaxf(mc, __shfl_xor(mc, 16));
    const float mnew  = fmaxf(mrun, mc);
    const float scale = __builtin_amdgcn_exp2f(mrun - mnew);
    mrun = mnew;
    float rs = 0.0f;
#pragma unroll
    for (int t = 0; t < 8; ++t)
#pragma unroll
      for (int r = 0; r < 8; ++r) {
        float p = __builtin_amdgcn_exp2f(st[t][r] - mnew);
        st[t][r] = p;
        rs += p;
      }
    rs += __shfl_xor(rs, 16);
    lrun = lrun * scale + rs;
#pragma unroll
    for (int dt = 0; dt < 16; ++dt)
#pragma unroll
      for (int r = 0; r < 8; ++r) ot[dt][r] *= scale;

    // ---- register-only P^T (C layout) -> 2nd-GEMM B fragments ----
    v16h pf[4];
#pragma unroll
    for (int t2 = 0; t2 < 4; ++t2) {
      BFrag bf;
#pragma unroll
      for (int j = 0; j < 4; ++j) {
        unsigned pe = packh2(st[2 * t2][2 * j],     st[2 * t2][2 * j + 1]);
        unsigned po = packh2(st[2 * t2 + 1][2 * j], st[2 * t2 + 1][2 * j + 1]);
        unsigned xe = (unsigned)__shfl_xor((int)pe, 16);
        unsigned xo = (unsigned)__shfl_xor((int)po, 16);
        bf.u[j]     = hi ? pe : pe;  // K = 2j,2j+1   (+8 for hi lanes)
        bf.u[4 + j] = hi ? po : po;  // K = 16+2j,... (+8 for hi lanes)
      }
      pf[t2] = bf.h;
    }

    // ---- GEMM2: O^T[d][q] += V^T[d][n] * P^T[n][q] ---------------------
    // 64 steps (t2-major, dt inner); rotating 4-slot pipeline, distance 3.
    {
      AFrag va[4];
#pragma unroll
      for (int s = 0; s < 3; ++s)
        ld_frag(va[s], vlds + ((s & 15) * 16 + ln) * VROW_B +
                       (s >> 4) * 64 + hi * 16);
#pragma unroll
      for (int s = 0; s < 64; ++s) {
        if (s + 3 < 64) {
          int sn = s + 3;
          ld_frag(va[sn & 3], vlds + ((sn & 15) * 16 + ln) * VROW_B +
                              (sn >> 4) * 64 + hi * 16);
        }
        const int t2 = s >> 4, dt = s & 15;
        ot[dt] = __builtin_amdgcn_wmma_f32_16x16x32_f16(
            false, va[s & 3].h, false, pf[t2], (short)0, ot[dt], false, false);
      }
    }
    __syncthreads();  // all waves done reading buffer `cur` before refill
  }

  // ---- epilogue: normalize, transpose O^T via LDS, coalesced store ----
  const float linv = 1.0f / fmaxf(lrun, 1e-30f);
  float* ost = (float*)smem;  // [128][OSTRIDE] staging (buffers now dead)
#pragma unroll
  for (int dt = 0; dt < 16; ++dt)
#pragma unroll
    for (int r = 0; r < 8; ++r) {
      int d = dt * 16 + r + hi * 8;   // C layout: VGPR r -> row d (+8 hi)
      ost[(wave * 16 + ln) * OSTRIDE + d] = ot[dt][r] * linv;
    }
  __syncthreads();
  {
    const int row = tid >> 1;
    const int cb  = (tid & 1) * 128;
    float* gout = out + ((size_t)blockIdx.x * BM + row) * D_DIM + cb;
    const float* src = ost + row * OSTRIDE + cb;
#pragma unroll
    for (int i = 0; i < 32; ++i)
      *(F4*)(gout + i * 4) = *(const F4*)(src + i * 4);
  }
}

// --------------------------------------------------------------------------
extern "C" void kernel_launch(void* const* d_in, const int* in_sizes, int n_in,
                              void* d_out, int out_size, void* d_ws,
                              size_t ws_size, hipStream_t stream) {
  const float* latent   = (const float*)d_in[0];  // [8,2048,256]
  const float* keys     = (const float*)d_in[1];  // [8192,256]
  const float* vals     = (const float*)d_in[2];  // [8192,256]
  const float* hardness = (const float*)d_in[3];  // [8192]
  float* out = (float*)d_out;

  char* ws = (char*)d_ws;
  _Float16* Q16 = (_Float16*)ws;
  _Float16* K16 = (_Float16*)(ws + (size_t)N_ROWS * D_DIM * 2);
  _Float16* Vt  = (_Float16*)(ws + (size_t)N_ROWS * D_DIM * 2 +
                              (size_t)M_KEYS * D_DIM * 2);

  // fold hardness * (1/TEMP) * log2(e) into the keys -> plain exp2 softmax
  const float KSCALE = 10.0f * 1.44269504088896340736f;

  rownorm_f16_kernel<<<N_ROWS, 256, 0, stream>>>(latent, nullptr, 1.0f, Q16);
  rownorm_f16_kernel<<<M_KEYS, 256, 0, stream>>>(keys, hardness, KSCALE, K16);
  transpose_vals_kernel<<<dim3(M_KEYS / 32, D_DIM / 32), dim3(32, 8), 0,
                          stream>>>(vals, Vt);

  (void)hipFuncSetAttribute(
      reinterpret_cast<const void*>(relmem_flash_kernel),
      hipFuncAttributeMaxDynamicSharedMemorySize, SMEM_MAIN);
  relmem_flash_kernel<<<N_ROWS / BM, 256, SMEM_MAIN, stream>>>(Q16, K16, Vt,
                                                               out);
}
